// HungarianMatcher_74139725463794
// MI455X (gfx1250) — compile-verified, matrix operand order checked by reading, not run on hardware
//
#include <hip/hip_runtime.h>
#include <math.h>

typedef __attribute__((ext_vector_type(16))) __bf16 v16bf;
typedef __attribute__((ext_vector_type(8)))  float  v8f;

#define BS   256
#define NQ   512
#define NC   128
#define NP   256
#define NG   128
#define BIGC 1.0e6f
#define EPSC 1e-6f
#define LN2F 0.69314718055994530942f

// ln(x) for x guaranteed normal (here x in [1e-6, 1]): bare v_log_f32, no
// subnormal pre-scale that __logf emits.
__device__ __forceinline__ float fast_ln_normal(float x) {
    return __builtin_amdgcn_logf(x) * LN2F;
}

// ---- Kernel 1: per-(b,p) softmax stats over Q (contiguous 512 floats/row) ----
// stats[row] = max_q + log(sum_q exp(x - max_q)); logP = x - stats
__global__ void colstats_kernel(const float* __restrict__ attn,
                                float* __restrict__ stats) {
    const int wave = threadIdx.x >> 5;
    const int lane = threadIdx.x & 31;
    const int row  = blockIdx.x * 8 + wave;              // [0, BS*NP)
    const float* p = attn + (size_t)row * NQ;
    float x[16];
    float mx = -INFINITY;
#pragma unroll
    for (int i = 0; i < 16; ++i) { x[i] = p[lane + 32 * i]; mx = fmaxf(mx, x[i]); }
#pragma unroll
    for (int off = 16; off >= 1; off >>= 1) mx = fmaxf(mx, __shfl_xor(mx, off, 32));
    float s = 0.f;
#pragma unroll
    for (int i = 0; i < 16; ++i) s += __expf(x[i] - mx);   // arg <= 0: safe fast path
#pragma unroll
    for (int off = 16; off >= 1; off >>= 1) s += __shfl_xor(s, off, 32);
    if (lane == 0) stats[row] = mx + __builtin_amdgcn_logf(s) * LN2F;  // s in [1,512]
}

// ---- Kernel 2: per-(b,q) logsumexp over C (contiguous 128 floats/row) ----
__global__ void actlse_kernel(const float* __restrict__ logits,
                              float* __restrict__ lse) {
    const int wave = threadIdx.x >> 5;
    const int lane = threadIdx.x & 31;
    const int row  = blockIdx.x * 8 + wave;              // [0, BS*NQ)
    const float* p = logits + (size_t)row * NC;
    float x[4];
    float mx = -INFINITY;
#pragma unroll
    for (int i = 0; i < 4; ++i) { x[i] = p[lane + 32 * i]; mx = fmaxf(mx, x[i]); }
#pragma unroll
    for (int off = 16; off >= 1; off >>= 1) mx = fmaxf(mx, __shfl_xor(mx, off, 32));
    float s = 0.f;
#pragma unroll
    for (int i = 0; i < 4; ++i) s += __expf(x[i] - mx);
#pragma unroll
    for (int off = 16; off >= 1; off >>= 1) s += __shfl_xor(s, off, 32);
    if (lane == 0) lse[row] = mx + __builtin_amdgcn_logf(s) * LN2F;    // s in [1,128]
}

// ---- Kernel 3: cost assembly; one-hot einsums as bf16 WMMA (hi/lo split) ----
// Block: (batch b, 64-query tile). Wave w: q-subtile (w&3), g-strip (w>>2).
__global__ __launch_bounds__(256)
void cost_kernel(const float* __restrict__ act_logits,
                 const float* __restrict__ attn,
                 const int* __restrict__ act_ids,
                 const int* __restrict__ grp_ids,
                 const float* __restrict__ colstats,
                 const float* __restrict__ act_lse,
                 float* __restrict__ out) {
    __shared__ float ms_s[NP];
    __shared__ int   gid_s[NP];
    __shared__ int   aid_s[NG];
    __shared__ int   mcnt_s[NG];
    __shared__ float rowtot_s[64];
    __shared__ float psize_s[64];
    __shared__ float lse_s[64];

    const int b    = blockIdx.x >> 3;
    const int q0   = (blockIdx.x & 7) * 64;
    const int t    = threadIdx.x;
    const int wave = t >> 5, lane = t & 31;
    const int n    = lane & 15;
    const bool lo  = lane < 16;
    const int qsub   = wave & 3;
    const int gstrip = wave >> 2;

    if (t < NG) { aid_s[t] = act_ids[b * NG + t]; mcnt_s[t] = 0; }
    ms_s[t] = colstats[b * NP + t];
    const int mygid = grp_ids[b * NP + t];
    gid_s[t] = mygid;
    if (t < 64) lse_s[t] = act_lse[(size_t)b * NQ + q0 + t];
    __syncthreads();
    atomicAdd(&mcnt_s[mygid], 1);        // completed before the post-loop barrier

    v8f accm[4], accn[4];
    const v8f z8 = {0.f, 0.f, 0.f, 0.f, 0.f, 0.f, 0.f, 0.f};
#pragma unroll
    for (int i = 0; i < 4; ++i) { accm[i] = z8; accn[i] = z8; }

    // A fragment layout (16-bit 16x32): lanes 0-15: M=lane, K in [0,8)+[16,24);
    // lanes 16-31: M=lane-16, K in [8,16)+[24,32). B mirrors with N=lane&15.
    const float* attn_b = attn + (size_t)b * NP * NQ + (q0 + qsub * 16 + n);
    const int khalf = lo ? 0 : 8;
    const int gbase = gstrip * 64 + n;

    float rt_ln = 0.f, rt_ps = 0.f;

    for (int kc = 0; kc < 8; ++kc) {          // P=256 in K-chunks of 32
        const int pbase = kc * 32;
        v16bf aPh, aPl, aNh, aNl;
        int gid16[16];
#pragma unroll
        for (int j = 0; j < 8; ++j) {
            const int kb = ((j < 4) ? 2 * j : 16 + 2 * (j - 4)) + khalf;
            const int p0 = pbase + kb, p1 = p0 + 1;
            const float x0 = attn_b[(size_t)p0 * NQ];
            const float x1 = attn_b[(size_t)p1 * NQ];
            const float lp0 = x0 - ms_s[p0];
            const float lp1 = x1 - ms_s[p1];
            // lp <= 0  =>  exp in (0,1]: fast exp has no edge cases to handle
            const float e0 = __expf(lp0), e1 = __expf(lp1);
            rt_ps += e0 + e1;
            const float c0 = fminf(fmaxf(e0, EPSC), 1.f - EPSC);
            const float c1 = fminf(fmaxf(e1, EPSC), 1.f - EPSC);
            // log1p(-c) with c in [1e-6, 1-1e-6]: 1-c in [1e-6,1] is normal
            const float ln0 = fast_ln_normal(1.f - c0);
            const float ln1 = fast_ln_normal(1.f - c1);
            rt_ln += ln0 + ln1;
            const __bf16 ph0 = (__bf16)lp0, ph1 = (__bf16)lp1;
            const __bf16 nh0 = (__bf16)ln0, nh1 = (__bf16)ln1;
            aPh[2 * j] = ph0;  aPh[2 * j + 1] = ph1;
            aNh[2 * j] = nh0;  aNh[2 * j + 1] = nh1;
            aPl[2 * j] = (__bf16)(lp0 - (float)ph0);
            aPl[2 * j + 1] = (__bf16)(lp1 - (float)ph1);
            aNl[2 * j] = (__bf16)(ln0 - (float)nh0);
            aNl[2 * j + 1] = (__bf16)(ln1 - (float)nh1);
            gid16[2 * j] = gid_s[p0];  gid16[2 * j + 1] = gid_s[p1];
        }
#pragma unroll
        for (int gt = 0; gt < 4; ++gt) {
            const int g = gbase + gt * 16;
            v16bf bM;
#pragma unroll
            for (int s2 = 0; s2 < 16; ++s2)
                bM[s2] = (gid16[s2] == g) ? (__bf16)1.0f : (__bf16)0.0f;
            // hi/lo split: two chained WMMAs per product -> ~f32 accuracy
            accm[gt] = __builtin_amdgcn_wmma_f32_16x16x32_bf16(
                false, aPl, false, bM, (short)0, accm[gt], false, false);
            accm[gt] = __builtin_amdgcn_wmma_f32_16x16x32_bf16(
                false, aPh, false, bM, (short)0, accm[gt], false, false);
            accn[gt] = __builtin_amdgcn_wmma_f32_16x16x32_bf16(
                false, aNl, false, bM, (short)0, accn[gt], false, false);
            accn[gt] = __builtin_amdgcn_wmma_f32_16x16x32_bf16(
                false, aNh, false, bM, (short)0, accn[gt], false, false);
        }
    }

    // Per-q totals: lane L and L+16 cover complementary K halves of row q=L.
    const float full_ln = rt_ln + __shfl_xor(rt_ln, 16, 32);
    const float full_ps = rt_ps + __shfl_xor(rt_ps, 16, 32);
    if (gstrip == 0 && lo) {
        rowtot_s[qsub * 16 + n] = full_ln;
        psize_s[qsub * 16 + n]  = full_ps;
    }
    __syncthreads();

    // C/D layout: lane holds column g = gbase+gt*16+n; VGPR r holds row r (lo)
    // or r+8 (hi half).
    const float* actrow = act_logits + ((size_t)b * NQ + q0) * NC;
#pragma unroll
    for (int gt = 0; gt < 4; ++gt) {
        const int g = gbase + gt * 16;
        const float m    = (float)mcnt_s[g];
        const float nmem = (float)NP - m;
        const int   aid  = aid_s[g];
#pragma unroll
        for (int r = 0; r < 8; ++r) {
            const int ql = qsub * 16 + (lo ? r : r + 8);
            const float mem  = accm[gt][r];
            const float mneg = accn[gt][r];
            const float non  = rowtot_s[ql] - mneg;
            float gcost = -mem / fmaxf(m, 1.f) - non / fmaxf(nmem, 1.f);
            float scost = fabsf(psize_s[ql] - m) * (1.0f / (float)NP);
            if (m == 0.f) { gcost = BIGC; scost = BIGC; }
            const float acost = lse_s[ql] - actrow[(size_t)ql * NC + aid];
            out[((size_t)b * NQ + q0 + ql) * NG + g] = gcost + acost + scost;
        }
    }
}

extern "C" void kernel_launch(void* const* d_in, const int* in_sizes, int n_in,
                              void* d_out, int out_size, void* d_ws, size_t ws_size,
                              hipStream_t stream) {
    (void)in_sizes; (void)n_in; (void)out_size; (void)ws_size;
    const float* act_logits = (const float*)d_in[0];   // [bs, Q, C]
    const float* attn       = (const float*)d_in[1];   // [bs, P, Q]
    const int*   act_ids    = (const int*)d_in[2];     // [bs, G]
    const int*   grp_ids    = (const int*)d_in[3];     // [bs, P]
    float* colstats = (float*)d_ws;                    // BS*NP floats (256 KB)
    float* act_lse  = colstats + BS * NP;              // BS*NQ floats (512 KB)
    float* out = (float*)d_out;

    colstats_kernel<<<(BS * NP) / 8, 256, 0, stream>>>(attn, colstats);
    actlse_kernel<<<(BS * NQ) / 8, 256, 0, stream>>>(act_logits, act_lse);
    cost_kernel<<<BS * 8, 256, 0, stream>>>(act_logits, attn, act_ids, grp_ids,
                                            colstats, act_lse, out);
}
